// TDMaskedAttention_59742995088113
// MI455X (gfx1250) — matrix operand order repaired and checked by reading, not compiled, not run on hardware
//
#include <hip/hip_runtime.h>

// ---------------- problem constants ----------------
#define B_    2
#define N_    2048
#define D_    512
#define H_    8
#define DH_   64
#define T_    16
#define BN_   (B_ * N_)          // 4096 rows
#define QKVC_ (3 * H_ * DH_)     // 1536
#define HDIM_ (H_ * DH_)         // 512
#define SCALE_ 0.125f            // DH^-0.5
#define EPS_  1e-5f

typedef __attribute__((ext_vector_type(16))) __bf16 v16bf;
typedef __attribute__((ext_vector_type(8)))  __bf16 v8bf;
typedef __attribute__((ext_vector_type(8)))  float  v8f;

// ---------------- bf16 helpers ----------------
__device__ __forceinline__ unsigned short f2bf(float f) {
  union { float f; unsigned u; } v; v.f = f;
  unsigned r = v.u + 0x7FFFu + ((v.u >> 16) & 1u);   // round-to-nearest-even
  return (unsigned short)(r >> 16);
}

// ---------------- WMMA wrapper ----------------
__device__ __forceinline__ v8f wmma_bf16(v16bf a, v16bf b, v8f c) {
  return __builtin_amdgcn_wmma_f32_16x16x32_bf16(
      false, a, false, b, (short)0, c, false, false);
}

// ---------------- vectorized fragment loaders ----------------
// 16 contiguous bf16 (32B, 32B-aligned) -> one v16bf (2x b128)
__device__ __forceinline__ v16bf load_contig16(const unsigned short* p) {
  return *(const v16bf*)p;
}
// A fragment: two 8-element chunks at +16*half and +16*half+8
// (contiguous k order matching B: lane=row, e -> K=e+16*half)
__device__ __forceinline__ v16bf load_a_pair(const unsigned short* p, int half) {
  v8bf lo = *(const v8bf*)(p + 16 * half);
  v8bf hi = *(const v8bf*)(p + 16 * half + 8);
  return __builtin_shufflevector(lo, hi, 0, 1, 2, 3, 4, 5, 6, 7,
                                         8, 9, 10, 11, 12, 13, 14, 15);
}

// ================= kernel 1: LayerNorm -> bf16 =================
__global__ __launch_bounds__(256)
void ln_kernel(const float* __restrict__ x, const float* __restrict__ g,
               const float* __restrict__ be, unsigned short* __restrict__ xn) {
  __shared__ float red[256];
  const int row = blockIdx.x, tid = threadIdx.x;
  const float* xr = x + (size_t)row * D_;
  const float v0 = xr[tid], v1 = xr[tid + 256];
  red[tid] = v0 + v1; __syncthreads();
#pragma unroll
  for (int o = 128; o > 0; o >>= 1) { if (tid < o) red[tid] += red[tid + o]; __syncthreads(); }
  const float mean = red[0] * (1.0f / D_);
  __syncthreads();
  const float d0 = v0 - mean, d1 = v1 - mean;
  red[tid] = d0 * d0 + d1 * d1; __syncthreads();
#pragma unroll
  for (int o = 128; o > 0; o >>= 1) { if (tid < o) red[tid] += red[tid + o]; __syncthreads(); }
  const float rstd = rsqrtf(red[0] * (1.0f / D_) + EPS_);
  unsigned short* xo = xn + (size_t)row * D_;
  xo[tid]       = f2bf(d0 * rstd * g[tid]       + be[tid]);
  xo[tid + 256] = f2bf(d1 * rstd * g[tid + 256] + be[tid + 256]);
}

// ================= kernel 2: weights fp32 -> bf16, TRANSPOSED =================
// dqT[c*D + k] = Wqkv[k*QKVC + c]; dwT[c*D + k] = Wout[k*D + c]
// -> B fragments become contiguous 32B loads per lane.
__global__ void conv_w_kernel(const float* __restrict__ wqkv, const float* __restrict__ wout,
                              unsigned short* __restrict__ dqT, unsigned short* __restrict__ dwT) {
  const int n1 = QKVC_ * D_, total = n1 + D_ * D_;
  for (int i = blockIdx.x * blockDim.x + threadIdx.x; i < total; i += gridDim.x * blockDim.x) {
    if (i < n1) {
      const int cc = i / D_, k = i - cc * D_;
      dqT[i] = f2bf(wqkv[(size_t)k * QKVC_ + cc]);
    } else {
      const int j = i - n1;
      const int cc = j / D_, k = j - cc * D_;
      dwT[j] = f2bf(wout[(size_t)k * D_ + cc]);
    }
  }
}

// ================= kernel 3: QKV GEMM (WMMA), 16x64 tile per wave =================
// q,k stored head-major [B*H][N][DH]; v stored TRANSPOSED [B*H][DH][N]
__global__ __launch_bounds__(128)
void qkv_gemm_kernel(const unsigned short* __restrict__ xn, const unsigned short* __restrict__ wT,
                     unsigned short* __restrict__ qb, unsigned short* __restrict__ kb,
                     unsigned short* __restrict__ vb) {
  const int lane = threadIdx.x & 31, wv = threadIdx.x >> 5;
  const int tile = blockIdx.x * 4 + wv;
  const int cg = tile % (QKVC_ / 64);          // 24 groups of 64 output cols
  const int rt = tile / (QKVC_ / 64);          // 256 row tiles
  const int half = lane >> 4, l15 = lane & 15;
  const int m = rt * 16 + l15;                 // A row for this lane

  v8f acc[4];
#pragma unroll
  for (int t = 0; t < 4; ++t) { v8f z = {}; acc[t] = z; }

  for (int kk = 0; kk < D_; kk += 32) {
    const v16bf af = load_a_pair(xn + (size_t)m * D_ + kk, half);
    __builtin_prefetch(xn + (size_t)m * D_ + kk + 32, 0, 3);
#pragma unroll
    for (int t = 0; t < 4; ++t) {
      const int n = cg * 64 + t * 16 + l15;
      const v16bf bf = load_contig16(wT + (size_t)n * D_ + kk + 16 * half);
      acc[t] = wmma_bf16(af, bf, acc[t]);
    }
  }
#pragma unroll
  for (int t = 0; t < 4; ++t) {
    const int col = cg * 64 + t * 16 + l15;
    const int sec = col >> 9;                  // 0=q 1=k 2=v (uniform per t)
    const int hd = col & 511, hh = hd >> 6, dh = hd & 63;
#pragma unroll
    for (int r = 0; r < 8; ++r) {
      const int rr = rt * 16 + r + 8 * half;
      const int b = rr >> 11, n = rr & (N_ - 1);
      const unsigned short val = f2bf(acc[t][r]);
      if (sec == 0)
        qb[((((size_t)b * H_) + hh) * N_ + n) * DH_ + dh] = val;
      else if (sec == 1)
        kb[((((size_t)b * H_) + hh) * N_ + n) * DH_ + dh] = val;
      else
        vb[((((size_t)b * H_) + hh) * DH_ + dh) * N_ + n] = val;   // transposed V
    }
  }
}

// ================= kernel 4: flash attention, one 16-row tile per wave =================
__global__ __launch_bounds__(128)
void attn_kernel(const unsigned short* __restrict__ qb, const unsigned short* __restrict__ kb,
                 const unsigned short* __restrict__ vb, const float* __restrict__ mask,
                 const float* __restrict__ R, const float* __restrict__ av,
                 const float* __restrict__ cv, unsigned short* __restrict__ aout) {
  __shared__ unsigned short ldsP[4][16 * 32];  // per-wave P-tile transpose buffer
  const int lane = threadIdx.x & 31, wv = threadIdx.x >> 5;
  const int tile = blockIdx.x * 4 + wv;
  const int qt = tile % (N_ / 16);
  const int bh = tile / (N_ / 16);
  const int b = bh / H_, h = bh % H_;
  const int row0 = qt * 16;
  const int half = lane >> 4, l15 = lane & 15;

  const unsigned short* Q  = qb + (size_t)bh * N_ * DH_;   // [N][DH]
  const unsigned short* K  = kb + (size_t)bh * N_ * DH_;   // [N][DH]
  const unsigned short* VT = vb + (size_t)bh * DH_ * N_;   // [DH][N]

  const v16bf qa0 = load_a_pair(Q + (size_t)(row0 + l15) * DH_ + 0, half);
  const v16bf qa1 = load_a_pair(Q + (size_t)(row0 + l15) * DH_ + 32, half);
  const float ah = fabsf(av[h]), ch = fabsf(cv[h]);
  const int trow = row0 >> 7;                  // 128 tokens per timepoint

  float mrow[8], lrow[8];
  v8f accO[4];
#pragma unroll
  for (int r = 0; r < 8; ++r) { mrow[r] = -1e30f; lrow[r] = 0.0f; }
#pragma unroll
  for (int ot = 0; ot < 4; ++ot) { v8f z = {}; accO[ot] = z; }

  unsigned short* pl = ldsP[wv];

  for (int j = 0; j < N_ / 32; ++j) {          // 64 key chunks of 32
    const int kc0 = j * 32;
    const float Rv = R[((size_t)b * T_ + trow) * T_ + (kc0 >> 7)];
    const float rd = 1.0f / (1.0f + __expf(ah * Rv - ch));  // sigmoid(|c|-|a|R)

    // ---- scores: two 16x16 tiles (B = K^T, contiguous per lane) ----
    const int n0 = kc0 + l15, n1 = kc0 + 16 + l15;
    v8f s0 = {}, s1 = {};
    s0 = wmma_bf16(qa0, load_contig16(K + (size_t)n0 * DH_ + 0  + 16 * half), s0);
    s0 = wmma_bf16(qa1, load_contig16(K + (size_t)n0 * DH_ + 32 + 16 * half), s0);
    s1 = wmma_bf16(qa0, load_contig16(K + (size_t)n1 * DH_ + 0  + 16 * half), s1);
    s1 = wmma_bf16(qa1, load_contig16(K + (size_t)n1 * DH_ + 32 + 16 * half), s1);

    // ---- relu + mask + scale + time decay, online softmax ----
    float sv0[8], sv1[8], p0[8], p1[8];
#pragma unroll
    for (int r = 0; r < 8; ++r) {
      const int row = row0 + r + 8 * half;
      const size_t mb = ((size_t)b * N_ + row) * N_ + kc0 + l15;
      sv0[r] = (fmaxf(s0[r], 0.0f) + mask[mb])      * SCALE_ * rd;
      sv1[r] = (fmaxf(s1[r], 0.0f) + mask[mb + 16]) * SCALE_ * rd;
    }
#pragma unroll
    for (int r = 0; r < 8; ++r) {
      float tm = fmaxf(sv0[r], sv1[r]);
#pragma unroll
      for (int off = 8; off >= 1; off >>= 1) tm = fmaxf(tm, __shfl_xor(tm, off));
      const float mn = fmaxf(mrow[r], tm);
      const float corr = __expf(mrow[r] - mn);
      p0[r] = __expf(sv0[r] - mn);
      p1[r] = __expf(sv1[r] - mn);
      float rs = p0[r] + p1[r];
#pragma unroll
      for (int off = 8; off >= 1; off >>= 1) rs += __shfl_xor(rs, off);
      lrow[r] = lrow[r] * corr + rs;
      mrow[r] = mn;
#pragma unroll
      for (int ot = 0; ot < 4; ++ot) accO[ot][r] *= corr;
      // stage P (C-layout) into LDS as row-major 16x32 bf16 tile
      pl[(r + 8 * half) * 32 + l15]      = f2bf(p0[r]);
      pl[(r + 8 * half) * 32 + 16 + l15] = f2bf(p1[r]);
    }
    // wave-private LDS: drain DS stores, re-read in A-fragment layout (2x ds_load_b128)
    asm volatile("s_wait_dscnt 0x0" ::: "memory");
    const v16bf pa = load_a_pair(pl + l15 * 32, half);
    asm volatile("" ::: "memory");

    // ---- O += P(16x32) * V(32x64), V transposed so B-frag is contiguous ----
#pragma unroll
    for (int ot = 0; ot < 4; ++ot) {
      const int dh = ot * 16 + l15;
      const v16bf bv = load_contig16(VT + (size_t)dh * N_ + kc0 + 16 * half);
      accO[ot] = wmma_bf16(pa, bv, accO[ot]);
    }
  }

  // ---- normalize and write concat-head output [B, N, H*DH] ----
#pragma unroll
  for (int ot = 0; ot < 4; ++ot)
#pragma unroll
    for (int r = 0; r < 8; ++r) {
      const int row = row0 + r + 8 * half;
      const float o = accO[ot][r] / lrow[r];
      aout[((size_t)b * N_ + row) * HDIM_ + h * DH_ + ot * 16 + l15] = f2bf(o);
    }
}

// ================= kernel 5: output GEMM (WMMA), 16x64 tile per wave =================
__global__ __launch_bounds__(128)
void out_gemm_kernel(const unsigned short* __restrict__ ao, const unsigned short* __restrict__ wT,
                     float* __restrict__ out) {
  const int lane = threadIdx.x & 31, wv = threadIdx.x >> 5;
  const int tile = blockIdx.x * 4 + wv;
  const int cg = tile % (D_ / 64);             // 8 groups of 64 output cols
  const int rt = tile / (D_ / 64);             // 256 row tiles
  const int half = lane >> 4, l15 = lane & 15;
  const int m = rt * 16 + l15;

  v8f acc[4];
#pragma unroll
  for (int t = 0; t < 4; ++t) { v8f z = {}; acc[t] = z; }

  for (int kk = 0; kk < HDIM_; kk += 32) {
    const v16bf af = load_a_pair(ao + (size_t)m * HDIM_ + kk, half);
    __builtin_prefetch(ao + (size_t)m * HDIM_ + kk + 32, 0, 3);
#pragma unroll
    for (int t = 0; t < 4; ++t) {
      const int n = cg * 64 + t * 16 + l15;
      const v16bf bf = load_contig16(wT + (size_t)n * D_ + kk + 16 * half);
      acc[t] = wmma_bf16(af, bf, acc[t]);
    }
  }
#pragma unroll
  for (int t = 0; t < 4; ++t)
#pragma unroll
    for (int r = 0; r < 8; ++r) {
      const int row = rt * 16 + r + 8 * half;
      out[(size_t)row * D_ + cg * 64 + t * 16 + l15] = acc[t][r];
    }
}

// ================= host launcher =================
extern "C" void kernel_launch(void* const* d_in, const int* in_sizes, int n_in,
                              void* d_out, int out_size, void* d_ws, size_t ws_size,
                              hipStream_t stream) {
  const float* x     = (const float*)d_in[0];
  const float* m     = (const float*)d_in[1];
  const float* R     = (const float*)d_in[2];
  const float* gamma = (const float*)d_in[3];
  const float* beta  = (const float*)d_in[4];
  const float* Wqkv  = (const float*)d_in[5];
  const float* Wout  = (const float*)d_in[6];
  const float* a     = (const float*)d_in[7];
  const float* c     = (const float*)d_in[8];

  unsigned short* xn  = (unsigned short*)d_ws;                   // [BN, D]
  unsigned short* wqT = xn  + (size_t)BN_ * D_;                  // [3HD, D] transposed
  unsigned short* woT = wqT + (size_t)QKVC_ * D_;                // [D, HD] transposed
  unsigned short* qb  = woT + (size_t)D_ * D_;                   // [B*H, N, DH]
  unsigned short* kb  = qb  + (size_t)B_ * H_ * N_ * DH_;        // [B*H, N, DH]
  unsigned short* vb  = kb  + (size_t)B_ * H_ * N_ * DH_;        // [B*H, DH, N] transposed
  unsigned short* ao  = vb  + (size_t)B_ * H_ * N_ * DH_;        // [BN, HD]

  ln_kernel<<<BN_, 256, 0, stream>>>(x, gamma, beta, xn);
  conv_w_kernel<<<1024, 256, 0, stream>>>(Wqkv, Wout, wqT, woT);
  qkv_gemm_kernel<<<(BN_ / 16) * (QKVC_ / 64) / 4, 128, 0, stream>>>(xn, wqT, qb, kb, vb);
  attn_kernel<<<B_ * H_ * (N_ / 16) / 4, 128, 0, stream>>>(qb, kb, vb, m, R, a, c, ao);
  out_gemm_kernel<<<(BN_ / 16) * (D_ / 64) / 4, 128, 0, stream>>>(ao, woT, (float*)d_out);
}
